// ImprovedGNNBinaryClassificationModel_60206851555950
// MI455X (gfx1250) — compile-verified, matrix operand order checked by reading, not run on hardware
//
#include <hip/hip_runtime.h>
#include <hip/hip_bf16.h>
#include <math.h>

// ---------------- types ----------------
typedef __attribute__((ext_vector_type(16))) __bf16 v16bf;
typedef __attribute__((ext_vector_type(8)))  float  v8f;

#define HIDC 64
#define NHEADS 2
#define HCC 128
#define NEG_SLOPE 0.2f
#define BN_EPS 1e-5f

// ---------------- device helpers ----------------
static __device__ __forceinline__ unsigned short f2bf(float f) {
  unsigned u = __float_as_uint(f);
  u += 0x7FFFu + ((u >> 16) & 1u);           // round-to-nearest-even
  return (unsigned short)(u >> 16);
}

static __device__ __forceinline__ void atomicMaxF(float* addr, float v) {
  if (v >= 0.0f) atomicMax((int*)addr, __float_as_int(v));
  else           atomicMin((unsigned int*)addr, __float_as_uint(v));
}

// ---------------- staging conversion kernels ----------------
__global__ void k_convert_pad(const float* __restrict__ x, unsigned short* __restrict__ o,
                              int M, int Mpad, int K) {
  long i = (long)blockIdx.x * blockDim.x + threadIdx.x;
  if (i >= (long)Mpad * K) return;
  int r = (int)(i / K);
  o[i] = (r < M) ? f2bf(x[i]) : (unsigned short)0;
}

// W is [K,N] row-major; output is Bt[N,K] bf16 (contiguous in K)
__global__ void k_transpose_convert(const float* __restrict__ W, unsigned short* __restrict__ o,
                                    int K, int N) {
  int i = blockIdx.x * blockDim.x + threadIdx.x;
  if (i >= K * N) return;
  int n = i / K, k = i - n * K;
  o[i] = f2bf(W[(long)k * N + n]);
}

__global__ void k_fill(float* __restrict__ p, float v, long n) {
  long i = (long)blockIdx.x * blockDim.x + threadIdx.x;
  if (i < n) p[i] = v;
}

// ---------------- WMMA GEMM: C[M,N] = A[M,K] x B[K,N] (+bias, opt relu) ----
// A: bf16 [M,K] row-major. Bt: bf16 [N,K] (i.e. B transposed). K % 32 == 0,
// M % 16 == 0, N % 16 == 0.
static __device__ __forceinline__ v16bf load_frag(const unsigned short* base, int ld, int lane) {
  int r  = lane & 15;
  int kb = (lane >> 4) << 3;                 // 0 or 8
  const unsigned short* p = base + (long)r * ld + kb;
  union { uint4 u[2]; v16bf v; } t;
  t.u[0] = *(const uint4*)(p);               // K = kb .. kb+7
  t.u[1] = *(const uint4*)(p + 16);          // K = 16+kb .. 16+kb+7
  return t.v;
}

__global__ __launch_bounds__(256)
void k_gemm_wmma(const unsigned short* __restrict__ A, const unsigned short* __restrict__ Bt,
                 const float* __restrict__ bias, float* __restrict__ C,
                 int M, int N, int K, int act) {
  int wid  = (blockIdx.x * blockDim.x + threadIdx.x) >> 5;   // wave-uniform
  int lane = threadIdx.x & 31;
  int ntiles = N >> 4;
  int tn = wid % ntiles;
  int tm = wid / ntiles;
  if (tm * 16 >= M) return;                  // uniform per wave -> EXEC stays all-1s
  const unsigned short* Abase = A + (long)tm * 16 * K;
  const unsigned short* Bbase = Bt + (long)tn * 16 * K;
  v8f acc = {};
  for (int k0 = 0; k0 < K; k0 += 32) {
    v16bf a = load_frag(Abase + k0, K, lane);
    v16bf b = load_frag(Bbase + k0, K, lane);
    acc = __builtin_amdgcn_wmma_f32_16x16x32_bf16(false, a, false, b,
                                                  (short)0, acc, false, false);
  }
  int col   = lane & 15;
  int rbase = (lane >> 4) << 3;
  int gc    = tn * 16 + col;
  float bv  = bias ? bias[gc] : 0.0f;
#pragma unroll
  for (int r = 0; r < 8; ++r) {
    float v = acc[r] + bv;
    if (act == 1) v = fmaxf(v, 0.0f);
    C[(long)(tm * 16 + rbase + r) * N + gc] = v;
  }
}

// ---------------- GAT attention kernels ----------------
// s[n,h] = sum_c h[n,h*64+c]*a_src[h,c] ; d likewise
__global__ void k_sd(const float* __restrict__ h, const float* __restrict__ a_src,
                     const float* __restrict__ a_dst,
                     float* __restrict__ s, float* __restrict__ d, int N) {
  int t = blockIdx.x * blockDim.x + threadIdx.x;
  if (t >= N * NHEADS) return;
  int n = t >> 1, hh = t & 1;
  const float* hp = h + (long)n * HCC + hh * HIDC;
  const float* as = a_src + hh * HIDC;
  const float* ad = a_dst + hh * HIDC;
  float ss = 0.f, dd = 0.f;
#pragma unroll 4
  for (int c = 0; c < HIDC; ++c) { float v = hp[c]; ss += v * as[c]; dd += v * ad[c]; }
  s[t] = ss; d[t] = dd;
}

// edge_index flat [2,E]; self loops appended: e in [E, E+N) -> src=dst=e-E
__global__ void k_alpha(const int* __restrict__ ei, const float* __restrict__ s,
                        const float* __restrict__ d, float* __restrict__ alpha,
                        float* __restrict__ amax, int E, int N) {
  int e = blockIdx.x * blockDim.x + threadIdx.x;
  if (e >= E + N) return;
  int srcI = (e < E) ? ei[e]     : (e - E);
  int dstI = (e < E) ? ei[E + e] : (e - E);
#pragma unroll
  for (int hh = 0; hh < NHEADS; ++hh) {
    float a = s[srcI * 2 + hh] + d[dstI * 2 + hh];
    a = (a > 0.f) ? a : NEG_SLOPE * a;       // leaky_relu
    alpha[(long)e * 2 + hh] = a;
    atomicMaxF(&amax[dstI * 2 + hh], a);
  }
}

__global__ void k_expsum(const int* __restrict__ ei, const float* __restrict__ amax,
                         float* __restrict__ alpha, float* __restrict__ den, int E, int N) {
  int e = blockIdx.x * blockDim.x + threadIdx.x;
  if (e >= E + N) return;
  int dstI = (e < E) ? ei[E + e] : (e - E);
#pragma unroll
  for (int hh = 0; hh < NHEADS; ++hh) {
    float ex = __expf(alpha[(long)e * 2 + hh] - amax[dstI * 2 + hh]);
    alpha[(long)e * 2 + hh] = ex;
    atomicAdd(&den[dstI * 2 + hh], ex);
  }
}

// 4 threads per edge, 32 columns each (columns 0-63 head0, 64-127 head1)
__global__ void k_aggregate(const int* __restrict__ ei, const float* __restrict__ alpha,
                            const float* __restrict__ den, const float* __restrict__ h,
                            float* __restrict__ agg, int E, int N) {
  long t = (long)blockIdx.x * blockDim.x + threadIdx.x;
  long e = t >> 2;
  if (e >= (long)E + N) return;
  int part = (int)(t & 3);
  int srcI = (e < E) ? ei[e]     : (int)(e - E);
  int dstI = (e < E) ? ei[E + e] : (int)(e - E);
  int c0 = part * 32;
  int hh = c0 >> 6;
  float coef = alpha[e * 2 + hh] / den[dstI * 2 + hh];
  const float* hs = h + (long)srcI * HCC + c0;
  float* ap = agg + (long)dstI * HCC + c0;
#pragma unroll 8
  for (int c = 0; c < 32; ++c) atomicAdd(&ap[c], coef * hs[c]);
}

__global__ void k_bias_act(float* __restrict__ x, const float* __restrict__ b,
                           int N, int C, int act) {
  long i = (long)blockIdx.x * blockDim.x + threadIdx.x;
  if (i >= (long)N * C) return;
  int c = (int)(i % C);
  float v = x[i] + b[c];
  if (act == 1) v = fmaxf(v, 0.f);
  x[i] = v;
}

// ---------------- BatchNorm ----------------
__global__ __launch_bounds__(256)
void k_bn_stats(const float* __restrict__ x, float* __restrict__ mean,
                float* __restrict__ rstd, int N, int C) {
  int c = blockIdx.x;
  float s = 0.f, s2 = 0.f;
  for (int r = threadIdx.x; r < N; r += blockDim.x) {
    float v = x[(long)r * C + c]; s += v; s2 += v * v;
  }
  __shared__ float sh[256], sh2[256];
  sh[threadIdx.x] = s; sh2[threadIdx.x] = s2;
  __syncthreads();
  for (int st = 128; st > 0; st >>= 1) {
    if (threadIdx.x < st) { sh[threadIdx.x] += sh[threadIdx.x + st]; sh2[threadIdx.x] += sh2[threadIdx.x + st]; }
    __syncthreads();
  }
  if (threadIdx.x == 0) {
    float m = sh[0] / N;
    float v = sh2[0] / N - m * m;
    mean[c] = m;
    rstd[c] = rsqrtf(v + BN_EPS);
  }
}

__global__ void k_bn_apply(float* __restrict__ x, const float* __restrict__ mean,
                           const float* __restrict__ rstd, const float* __restrict__ g,
                           const float* __restrict__ b, int N, int C) {
  long i = (long)blockIdx.x * blockDim.x + threadIdx.x;
  if (i >= (long)N * C) return;
  int c = (int)(i % C);
  x[i] = (x[i] - mean[c]) * rstd[c] * g[c] + b[c];
}

// ---------------- head mean / pooling / readout ----------------
__global__ void k_head_mean(const float* __restrict__ x, float* __restrict__ o, int N) {
  long i = (long)blockIdx.x * blockDim.x + threadIdx.x;
  if (i >= (long)N * HIDC) return;
  long n = i / HIDC; int c = (int)(i % HIDC);
  o[i] = 0.5f * (x[n * HCC + 2 * c] + x[n * HCC + 2 * c + 1]);
}

__global__ void k_pool_acc(const float* __restrict__ x, const int* __restrict__ seg,
                           float* __restrict__ sum, float* __restrict__ cnt, int N, int C) {
  long i = (long)blockIdx.x * blockDim.x + threadIdx.x;
  if (i >= (long)N * C) return;
  int n = (int)(i / C), c = (int)(i % C);
  int g = seg[n];
  atomicAdd(&sum[(long)g * C + c], x[i]);
  if (c == 0) atomicAdd(&cnt[g], 1.0f);
}

__global__ void k_pool_div(float* __restrict__ sum, const float* __restrict__ cnt, int G, int C) {
  int i = blockIdx.x * blockDim.x + threadIdx.x;
  if (i >= G * C) return;
  sum[i] /= fmaxf(cnt[i / C], 1.0f);
}

__global__ void k_build_comb(const float* __restrict__ pn, const float* __restrict__ pe,
                             const float* __restrict__ wh, unsigned short* __restrict__ comb, int G) {
  int i = blockIdx.x * blockDim.x + threadIdx.x;
  if (i >= G * 192) return;
  int g = i / 192, c = i % 192;
  float v = (c < 64) ? pn[g * 64 + c] : (c < 128) ? pe[g * 64 + c - 64] : wh[g * 64 + c - 128];
  comb[i] = f2bf(v);
}

__global__ void k_final(const float* __restrict__ oh, const float* __restrict__ W2,
                        const float* __restrict__ b2, float* __restrict__ out, int G) {
  int g = blockIdx.x * blockDim.x + threadIdx.x;
  if (g >= G) return;
  float s = b2[0];
#pragma unroll 8
  for (int c = 0; c < 64; ++c) s += oh[g * 64 + c] * W2[c];
  out[g] = 1.0f / (1.0f + __expf(-s));
}

// ---------------- host orchestration ----------------
static inline int cdiv(long a, long b) { return (int)((a + b - 1) / b); }

struct GatBufs {
  unsigned short *bfA, *bfW;
  float *h, *agg, *s, *d, *amax, *den, *alpha, *mean, *rstd;
};

static void gat_layer(const float* x_in, int K, const float* W, const float* a_s,
                      const float* a_d, const float* bias, int relu_after,
                      const float* bn_g, const float* bn_b,
                      const int* ei, int E, int N, int Mpad,
                      const GatBufs& B, hipStream_t stream) {
  const int T = 256;
  k_convert_pad<<<cdiv((long)Mpad * K, T), T, 0, stream>>>(x_in, B.bfA, N, Mpad, K);
  k_transpose_convert<<<cdiv((long)K * HCC, T), T, 0, stream>>>(W, B.bfW, K, HCC);
  int waves = (Mpad / 16) * (HCC / 16);
  k_gemm_wmma<<<cdiv(waves, 8), T, 0, stream>>>(B.bfA, B.bfW, nullptr, B.h, Mpad, HCC, K, 0);
  k_sd<<<cdiv((long)N * NHEADS, T), T, 0, stream>>>(B.h, a_s, a_d, B.s, B.d, N);
  k_fill<<<cdiv((long)Mpad * 2, T), T, 0, stream>>>(B.amax, -3.0e38f, (long)Mpad * 2);
  k_fill<<<cdiv((long)Mpad * 2, T), T, 0, stream>>>(B.den, 0.0f, (long)Mpad * 2);
  k_fill<<<cdiv((long)Mpad * HCC, T), T, 0, stream>>>(B.agg, 0.0f, (long)Mpad * HCC);
  int Etot = E + N;
  k_alpha<<<cdiv(Etot, T), T, 0, stream>>>(ei, B.s, B.d, B.alpha, B.amax, E, N);
  k_expsum<<<cdiv(Etot, T), T, 0, stream>>>(ei, B.amax, B.alpha, B.den, E, N);
  k_aggregate<<<cdiv((long)Etot * 4, T), T, 0, stream>>>(ei, B.alpha, B.den, B.h, B.agg, E, N);
  k_bias_act<<<cdiv((long)N * HCC, T), T, 0, stream>>>(B.agg, bias, N, HCC, relu_after);
  k_bn_stats<<<HCC, T, 0, stream>>>(B.agg, B.mean, B.rstd, N, HCC);
  k_bn_apply<<<cdiv((long)N * HCC, T), T, 0, stream>>>(B.agg, B.mean, B.rstd, bn_g, bn_b, N, HCC);
}

extern "C" void kernel_launch(void* const* d_in, const int* in_sizes, int n_in,
                              void* d_out, int out_size, void* d_ws, size_t ws_size,
                              hipStream_t stream) {
  // input layout: 3 tensors, 34 params (dict order), edge_index, node_batch, edge_batch
  const float* node_x  = (const float*)d_in[0];
  const float* edge_x  = (const float*)d_in[1];
  const float* wx      = (const float*)d_in[2];
  const float** P = (const float**)(d_in + 3);
  const float *ng1_W=P[0],*ng1_as=P[1],*ng1_ad=P[2],*ng1_b=P[3];
  const float *ng2_W=P[4],*ng2_as=P[5],*ng2_ad=P[6],*ng2_b=P[7];
  const float *eg1_W=P[8],*eg1_as=P[9],*eg1_ad=P[10],*eg1_b=P[11];
  const float *eg2_W=P[12],*eg2_as=P[13],*eg2_ad=P[14],*eg2_b=P[15];
  const float *nbn1_g=P[16],*nbn1_b=P[17],*nbn2_g=P[18],*nbn2_b=P[19];
  const float *ebn1_g=P[20],*ebn1_b=P[21],*ebn2_g=P[22],*ebn2_b=P[23];
  const float *wbn_g=P[24],*wbn_b=P[25];
  const float *wm1_W=P[26],*wm1_b=P[27],*wm2_W=P[28],*wm2_b=P[29];
  const float *ro1_W=P[30],*ro1_b=P[31],*ro2_W=P[32],*ro2_b=P[33];
  const int* edge_index = (const int*)d_in[37];
  const int* node_batch = (const int*)d_in[38];
  const int* edge_batch = (const int*)d_in[39];

  const int ND = 128, ED = 64, WD = 32;
  int N = in_sizes[0] / ND;
  int E = in_sizes[37] / 2;
  int G = in_sizes[2] / WD;           // 64
  int Mpad = (N + 15) & ~15;

  // workspace carving
  char* ws = (char*)d_ws; size_t off = 0;
  auto carve = [&](size_t bytes) -> void* {
    void* p = ws + off; off += (bytes + 255) & ~(size_t)255; return p;
  };
  GatBufs B;
  B.bfA  = (unsigned short*)carve((size_t)Mpad * HCC * 2);
  B.bfW  = (unsigned short*)carve((size_t)192 * 128 * 2);
  B.h    = (float*)carve((size_t)Mpad * HCC * 4);
  B.agg  = (float*)carve((size_t)Mpad * HCC * 4);
  B.s    = (float*)carve((size_t)Mpad * 2 * 4);
  B.d    = (float*)carve((size_t)Mpad * 2 * 4);
  B.amax = (float*)carve((size_t)Mpad * 2 * 4);
  B.den  = (float*)carve((size_t)Mpad * 2 * 4);
  B.alpha= (float*)carve((size_t)(E + N) * 2 * 4);
  B.mean = (float*)carve(128 * 4);
  B.rstd = (float*)carve(128 * 4);
  float* nhm   = (float*)carve((size_t)N * HIDC * 4);
  float* ehm   = (float*)carve((size_t)N * HIDC * 4);
  float* pn    = (float*)carve((size_t)G * HIDC * 4);
  float* pe    = (float*)carve((size_t)G * HIDC * 4);
  float* cntn  = (float*)carve((size_t)G * 4);
  float* cnte  = (float*)carve((size_t)G * 4);
  unsigned short* wbf    = (unsigned short*)carve((size_t)G * 64 * 2);
  float* wh1   = (float*)carve((size_t)G * 64 * 4);
  float* wh2   = (float*)carve((size_t)G * 64 * 4);
  unsigned short* combbf = (unsigned short*)carve((size_t)G * 192 * 2);
  float* oh    = (float*)carve((size_t)G * 64 * 4);

  const int T = 256;

  // ---- node GAT stack ----
  gat_layer(node_x, ND, ng1_W, ng1_as, ng1_ad, ng1_b, 1, nbn1_g, nbn1_b,
            edge_index, E, N, Mpad, B, stream);
  gat_layer(B.agg, HCC, ng2_W, ng2_as, ng2_ad, ng2_b, 0, nbn2_g, nbn2_b,
            edge_index, E, N, Mpad, B, stream);
  k_head_mean<<<cdiv((long)N * HIDC, T), T, 0, stream>>>(B.agg, nhm, N);

  // ---- edge GAT stack (edge_attr is node-indexed, same graph) ----
  gat_layer(edge_x, ED, eg1_W, eg1_as, eg1_ad, eg1_b, 1, ebn1_g, ebn1_b,
            edge_index, E, N, Mpad, B, stream);
  gat_layer(B.agg, HCC, eg2_W, eg2_as, eg2_ad, eg2_b, 0, ebn2_g, ebn2_b,
            edge_index, E, N, Mpad, B, stream);
  k_head_mean<<<cdiv((long)N * HIDC, T), T, 0, stream>>>(B.agg, ehm, N);

  // ---- weather MLP (G=64 rows -> 4x4 grid of 16x16 WMMA tiles) ----
  k_convert_pad<<<cdiv((long)G * WD, T), T, 0, stream>>>(wx, wbf, G, G, WD);
  k_transpose_convert<<<cdiv((long)WD * 64, T), T, 0, stream>>>(wm1_W, B.bfW, WD, 64);
  k_gemm_wmma<<<cdiv((G / 16) * 4, 8), T, 0, stream>>>(wbf, B.bfW, wm1_b, wh1, G, 64, WD, 1);
  k_bn_stats<<<64, T, 0, stream>>>(wh1, B.mean, B.rstd, G, 64);
  k_bn_apply<<<cdiv((long)G * 64, T), T, 0, stream>>>(wh1, B.mean, B.rstd, wbn_g, wbn_b, G, 64);
  k_convert_pad<<<cdiv((long)G * 64, T), T, 0, stream>>>(wh1, wbf, G, G, 64);
  k_transpose_convert<<<cdiv((long)64 * 64, T), T, 0, stream>>>(wm2_W, B.bfW, 64, 64);
  k_gemm_wmma<<<cdiv((G / 16) * 4, 8), T, 0, stream>>>(wbf, B.bfW, wm2_b, wh2, G, 64, 64, 0);

  // ---- pooling ----
  k_fill<<<cdiv((long)G * 64, T), T, 0, stream>>>(pn, 0.f, (long)G * 64);
  k_fill<<<cdiv((long)G * 64, T), T, 0, stream>>>(pe, 0.f, (long)G * 64);
  k_fill<<<1, 64, 0, stream>>>(cntn, 0.f, G);
  k_fill<<<1, 64, 0, stream>>>(cnte, 0.f, G);
  k_pool_acc<<<cdiv((long)N * 64, T), T, 0, stream>>>(nhm, node_batch, pn, cntn, N, 64);
  k_pool_acc<<<cdiv((long)N * 64, T), T, 0, stream>>>(ehm, edge_batch, pe, cnte, N, 64);
  k_pool_div<<<cdiv((long)G * 64, T), T, 0, stream>>>(pn, cntn, G, 64);
  k_pool_div<<<cdiv((long)G * 64, T), T, 0, stream>>>(pe, cnte, G, 64);

  // ---- readout: [G,192] @ [192,64] (WMMA) -> relu -> [G,64] @ [64,1] -> sigmoid ----
  k_build_comb<<<cdiv((long)G * 192, T), T, 0, stream>>>(pn, pe, wh2, combbf, G);
  k_transpose_convert<<<cdiv((long)192 * 64, T), T, 0, stream>>>(ro1_W, B.bfW, 192, 64);
  k_gemm_wmma<<<cdiv((G / 16) * 4, 8), T, 0, stream>>>(combbf, B.bfW, ro1_b, oh, G, 64, 192, 1);
  k_final<<<cdiv(G, 64), 64, 0, stream>>>(oh, ro2_W, ro2_b, (float*)d_out, G);
}
